// RetnetBlock_66460323938437
// MI455X (gfx1250) — matrix operand order repaired and checked by reading, not compiled, not run on hardware
//
#include <hip/hip_runtime.h>
#include <hip/hip_bf16.h>

// ---------------------------------------------------------------------------
// RetNet block for MI455X (gfx1250, wave32).
// - All GEMMs on v_wmma_f32_16x16x32_bf16 (bf16 in, f32 accum).
// - Weight GEMMs: direct-from-global fragments (weights ~26MB bf16, L2-resident).
// - Retention (q.kT * decay) @ v * silu(xo): fully fused flash-style kernel,
//   scores staged per 128x32 chunk in LDS (double buffered) -- avoids the
//   256MB scores round-trip and the ret f32 round-trip.
// ---------------------------------------------------------------------------

#define BB    4
#define LL    1024
#define DD    1024
#define NH    16
#define HDIM  64
#define DFFD  4096
#define MROWS (BB * LL)   // 4096

typedef __bf16 bf16_t;
typedef __attribute__((__ext_vector_type__(16))) __bf16 v16bf;
typedef __attribute__((__ext_vector_type__(8)))  float  v8f;

union Frag16 { v16bf v; uint4 u[2]; };

// --------------------------- WMMA fragment loads ---------------------------
// A (M x K row-major), 16-bit A layout (ISA 7.12.2):
//   lane l: m = l%16, half = l/16
//   VGPR 0..3 = K[half*8 .. +7], VGPR 4..7 = K[16+half*8 .. +7] (contiguous)
__device__ __forceinline__ v16bf load_a_frag(const bf16_t* __restrict__ A, int lda,
                                             int mBase, int kBase, int lane) {
  const int m = mBase + (lane & 15);
  const int half = lane >> 4;
  const bf16_t* row = A + (long long)m * lda + kBase + half * 8;
  Frag16 f;
  f.u[0] = *reinterpret_cast<const uint4*>(row);
  f.u[1] = *reinterpret_cast<const uint4*>(row + 16);
  return f.v;
}

// B (K x N), sourced from row-major W[N,K] (X @ W.T form):
//   lane l: n = l%16, kh = l/16; VGPR 0..3 = K[kh*8 .. +7], VGPR 4..7 = K[16+kh*8 .. +7] of row n
__device__ __forceinline__ v16bf load_b_frag(const bf16_t* __restrict__ W, int ldw,
                                             int nBase, int kBase, int lane) {
  const int n = nBase + (lane & 15);
  const int kh = lane >> 4;
  const bf16_t* row = W + (long long)n * ldw + kBase + kh * 8;
  Frag16 f;
  f.u[0] = *reinterpret_cast<const uint4*>(row);
  f.u[1] = *reinterpret_cast<const uint4*>(row + 16);
  return f.v;
}

__device__ __forceinline__ v8f wmma_bf16(v16bf a, v16bf b, v8f c) {
  return __builtin_amdgcn_wmma_f32_16x16x32_bf16(false, a, false, b, (short)0, c,
                                                 false, false);
}

// ------------------------------ generic GEMM -------------------------------
// C[M,Nn](f32) = A[M,K](bf16) @ W[Nn,K](bf16)^T + bias
// 256 thr = 8 waves; block tile 128(M) x 64(N); wave tile 32x32 (2x2 WMMA).
__global__ __launch_bounds__(256) void gemm_bf16_kernel(
    const bf16_t* __restrict__ A, const bf16_t* __restrict__ W,
    const float* __restrict__ bias, float* __restrict__ C,
    int K, int lda, int ldw, int ldc) {
  const int lane = threadIdx.x & 31;
  const int wave = threadIdx.x >> 5;
  const int mW = blockIdx.y * 128 + (wave & 3) * 32;
  const int nW = blockIdx.x * 64 + (wave >> 2) * 32;

  const v8f zero8 = {0.f, 0.f, 0.f, 0.f, 0.f, 0.f, 0.f, 0.f};
  v8f acc[2][2];
  acc[0][0] = zero8; acc[0][1] = zero8; acc[1][0] = zero8; acc[1][1] = zero8;

  for (int k = 0; k < K; k += 32) {
    v16bf a0 = load_a_frag(A, lda, mW,      k, lane);
    v16bf a1 = load_a_frag(A, lda, mW + 16, k, lane);
    v16bf b0 = load_b_frag(W, ldw, nW,      k, lane);
    v16bf b1 = load_b_frag(W, ldw, nW + 16, k, lane);
    acc[0][0] = wmma_bf16(a0, b0, acc[0][0]);
    acc[0][1] = wmma_bf16(a0, b1, acc[0][1]);
    acc[1][0] = wmma_bf16(a1, b0, acc[1][0]);
    acc[1][1] = wmma_bf16(a1, b1, acc[1][1]);
  }

  const int colLane = lane & 15;
  const int rowOff = (lane >> 4) * 8;
#pragma unroll
  for (int mi = 0; mi < 2; ++mi) {
#pragma unroll
    for (int ni = 0; ni < 2; ++ni) {
      const int nBase = nW + ni * 16;
      const float bval = bias ? bias[nBase + colLane] : 0.f;
#pragma unroll
      for (int i = 0; i < 8; ++i) {
        const int row = mW + mi * 16 + rowOff + i;
        C[(long long)row * ldc + nBase + colLane] = acc[mi][ni][i] + bval;
      }
    }
  }
}

// -------------------- fused retention (flash-style) ------------------------
// Per batch z = b*NH + n:
//   ret[l, hd] = sum_m ( (q[l,:]·k[m,:]) * decay[n,l,m] ) * v[m,hd]
// Block tile: 128(l) x 64(HD); loop m in chunks of 32.
// Phase 1: 8 waves compute the 128x32 score chunk (each wave 32(l)x16(m),
//          2x2 WMMAs over K=HD=64), multiply by bf16 decay, stage in LDS.
// Phase 2: same waves consume LDS chunk as A-fragments (ds_load_b128 does the
//          C-layout -> A-layout transposition) and accumulate S@v.
// Epilogue fuses the silu gate: retg = bf16(ret * siluxo).
__global__ __launch_bounds__(256) void retention_kernel(
    const bf16_t* __restrict__ qS, const bf16_t* __restrict__ kS,
    const bf16_t* __restrict__ vT, const bf16_t* __restrict__ decayBF,
    const float* __restrict__ xoS, bf16_t* __restrict__ retg) {
  __shared__ __align__(16) bf16_t sS[2][128][32];   // 16KB double buffer

  const int z = blockIdx.z;                 // b*NH + n
  const int b = z >> 4;
  const int n = z & (NH - 1);
  const bf16_t* q = qS + (long long)z * LL * HDIM;
  const bf16_t* k = kS + (long long)z * LL * HDIM;
  const bf16_t* v = vT + (long long)z * HDIM * LL;
  const bf16_t* dec = decayBF + (long long)n * LL * LL;
  const float* xo = xoS + (long long)b * LL * DD + n * HDIM;
  bf16_t* G = retg + (long long)b * LL * DD + n * HDIM;

  const int lane = threadIdx.x & 31;
  const int wave = threadIdx.x >> 5;
  const int waveL = wave & 3;               // l-quadrant within block tile
  const int waveH = wave >> 2;              // scores: m-half; AV: hd-half
  const int lW = blockIdx.y * 128 + waveL * 32;

  // q fragments are invariant over the m loop: hoist (2 l-tiles x 2 k-steps)
  v16bf qf00 = load_a_frag(q, HDIM, lW,      0,  lane);
  v16bf qf01 = load_a_frag(q, HDIM, lW,      32, lane);
  v16bf qf10 = load_a_frag(q, HDIM, lW + 16, 0,  lane);
  v16bf qf11 = load_a_frag(q, HDIM, lW + 16, 32, lane);

  const v8f zero8 = {0.f, 0.f, 0.f, 0.f, 0.f, 0.f, 0.f, 0.f};
  v8f acc[2][2];
  acc[0][0] = zero8; acc[0][1] = zero8; acc[1][0] = zero8; acc[1][1] = zero8;

  const int colLane = lane & 15;
  const int rowOff = (lane >> 4) * 8;

  for (int m = 0; m < LL; m += 32) {
    const int p = (m >> 5) & 1;
    // --- phase 1: scores chunk rows [lW,+32), cols [m+waveH*16,+16)
    const int mCol = m + waveH * 16;
    v16bf kb0 = load_b_frag(k, HDIM, mCol, 0,  lane);
    v16bf kb1 = load_b_frag(k, HDIM, mCol, 32, lane);
    v8f s0 = zero8, s1 = zero8;
    s0 = wmma_bf16(qf00, kb0, s0);
    s0 = wmma_bf16(qf01, kb1, s0);
    s1 = wmma_bf16(qf10, kb0, s1);
    s1 = wmma_bf16(qf11, kb1, s1);

    const int lc = waveH * 16 + colLane;    // local score column 0..31
#pragma unroll
    for (int i = 0; i < 8; ++i) {
      const int lr0 = waveL * 32 + rowOff + i;        // local row 0..127
      const long long g0 = (long long)(lW + rowOff + i) * LL + mCol + colLane;
      sS[p][lr0][lc]      = (bf16_t)(s0[i] * (float)dec[g0]);
      sS[p][lr0 + 16][lc] = (bf16_t)(s1[i] * (float)dec[g0 + 16LL * LL]);
    }
    __syncthreads();

    // --- phase 2: ret[lW..+32, waveH*32..+32] += S_chunk @ v_chunk
    const int hdW = waveH * 32;
    v16bf sa0 = load_a_frag(&sS[p][0][0], 32, waveL * 32,      0, lane);
    v16bf sa1 = load_a_frag(&sS[p][0][0], 32, waveL * 32 + 16, 0, lane);
    v16bf vb0 = load_b_frag(v, LL, hdW,      m, lane);
    v16bf vb1 = load_b_frag(v, LL, hdW + 16, m, lane);
    acc[0][0] = wmma_bf16(sa0, vb0, acc[0][0]);
    acc[0][1] = wmma_bf16(sa0, vb1, acc[0][1]);
    acc[1][0] = wmma_bf16(sa1, vb0, acc[1][0]);
    acc[1][1] = wmma_bf16(sa1, vb1, acc[1][1]);
  }

  // --- epilogue: fused silu gate, bf16 store in [B,L,D] order
#pragma unroll
  for (int mi = 0; mi < 2; ++mi) {
#pragma unroll
    for (int ni = 0; ni < 2; ++ni) {
      const int hdBase = (wave >> 2) * 32 + ni * 16;
#pragma unroll
      for (int i = 0; i < 8; ++i) {
        const int row = lW + mi * 16 + rowOff + i;
        const long long idx = (long long)row * DD + hdBase + colLane;
        G[idx] = (bf16_t)(acc[mi][ni][i] * xo[idx]);
      }
    }
  }
}

// ------------------------------ LayerNorm ----------------------------------
__global__ __launch_bounds__(256) void ln_bf16_kernel(
    const float* __restrict__ x, const float* __restrict__ g,
    const float* __restrict__ b, bf16_t* __restrict__ out) {
  const int row = blockIdx.x;
  const int t = threadIdx.x;
  const float* xr = x + (long long)row * DD;
  float4 v4 = reinterpret_cast<const float4*>(xr)[t];
  float vv[4] = {v4.x, v4.y, v4.z, v4.w};
  float s = vv[0] + vv[1] + vv[2] + vv[3];
  float s2 = vv[0] * vv[0] + vv[1] * vv[1] + vv[2] * vv[2] + vv[3] * vv[3];

  __shared__ float red[256];
  __shared__ float red2[256];
  red[t] = s;
  red2[t] = s2;
  __syncthreads();
  for (int off = 128; off > 0; off >>= 1) {
    if (t < off) { red[t] += red[t + off]; red2[t] += red2[t + off]; }
    __syncthreads();
  }
  const float mean = red[0] * (1.0f / DD);
  const float var = red2[0] * (1.0f / DD) - mean * mean;
  const float rs = rsqrtf(var + 1e-5f);

  const int base = t * 4;
#pragma unroll
  for (int j = 0; j < 4; ++j) {
    const float y = (vv[j] - mean) * rs * g[base + j] + b[base + j];
    out[(long long)row * DD + base + j] = (bf16_t)y;
  }
}

// ------------------------------ elementwise --------------------------------
__global__ void cvt_bf16_kernel(const float* __restrict__ src,
                                bf16_t* __restrict__ dst, long long n) {
  long long i = (long long)blockIdx.x * blockDim.x + threadIdx.x;
  const long long stride = (long long)gridDim.x * blockDim.x;
  for (; i < n; i += stride) dst[i] = (bf16_t)src[i];
}

// p[B,L,4D] -> q_s[B,N,L,HD], k_s[B,N,L,HD], vT[B,N,HD,L] (bf16), silu(xo) f32
__global__ void split_proj_kernel(const float* __restrict__ p,
                                  bf16_t* __restrict__ qS, bf16_t* __restrict__ kS,
                                  bf16_t* __restrict__ vT, float* __restrict__ xoS) {
  long long i = (long long)blockIdx.x * blockDim.x + threadIdx.x;
  const long long total = (long long)BB * LL * DD;
  const long long stride = (long long)gridDim.x * blockDim.x;
  for (; i < total; i += stride) {
    const int dd = (int)(i % DD);
    const long long bl = i / DD;
    const int l = (int)(bl % LL);
    const int b = (int)(bl / LL);
    const int n = dd / HDIM;
    const int d = dd - n * HDIM;
    const float* pr = p + bl * (4LL * DD);
    const float qv = pr[dd];
    const float kv = pr[DD + dd];
    const float vv = pr[2 * DD + dd];
    const float xv = pr[3 * DD + dd];
    const long long head = (long long)(b * NH + n);
    qS[(head * LL + l) * HDIM + d] = (bf16_t)qv;
    kS[(head * LL + l) * HDIM + d] = (bf16_t)kv;
    vT[(head * HDIM + d) * LL + l] = (bf16_t)vv;
    xoS[i] = xv * (1.0f / (1.0f + expf(-xv)));  // silu applied once here
  }
}

__global__ void add_f32_kernel(const float* __restrict__ a,
                               const float* __restrict__ b,
                               float* __restrict__ o, long long n) {
  long long i = (long long)blockIdx.x * blockDim.x + threadIdx.x;
  const long long stride = (long long)gridDim.x * blockDim.x;
  for (; i < n; i += stride) o[i] = a[i] + b[i];
}

__global__ void gelu_bf16_kernel(const float* __restrict__ src,
                                 bf16_t* __restrict__ dst, long long n) {
  long long i = (long long)blockIdx.x * blockDim.x + threadIdx.x;
  const long long stride = (long long)gridDim.x * blockDim.x;
  for (; i < n; i += stride) {
    const float v = src[i];
    dst[i] = (bf16_t)(0.5f * v * (1.0f + erff(v * 0.70710678118654752f)));
  }
}

// ------------------------------- launcher ----------------------------------
extern "C" void kernel_launch(void* const* d_in, const int* in_sizes, int n_in,
                              void* d_out, int out_size, void* d_ws, size_t ws_size,
                              hipStream_t stream) {
  const float* x      = (const float*)d_in[0];
  const float* decay  = (const float*)d_in[1];
  const float* ln1_g  = (const float*)d_in[2];
  const float* ln1_b  = (const float*)d_in[3];
  const float* proj_W = (const float*)d_in[4];
  const float* proj_b = (const float*)d_in[5];
  const float* reto_W = (const float*)d_in[6];
  const float* reto_b = (const float*)d_in[7];
  const float* ln2_g  = (const float*)d_in[8];
  const float* ln2_b  = (const float*)d_in[9];
  const float* ffn1_W = (const float*)d_in[10];
  const float* ffn1_b = (const float*)d_in[11];
  const float* ffn2_W = (const float*)d_in[12];
  const float* ffn2_b = (const float*)d_in[13];
  float* out = (float*)d_out;

  char* wsBase = (char*)d_ws;
  size_t off = 0;
  auto carve = [&](size_t bytes) -> void* {
    void* p = wsBase + off;
    off = (off + bytes + 255) & ~(size_t)255;
    return p;
  };

  const long long BLD = (long long)BB * LL * DD;  // 4M
  bf16_t* wProj  = (bf16_t*)carve((size_t)4 * DD * DD * 2);
  bf16_t* wReto  = (bf16_t*)carve((size_t)DD * DD * 2);
  bf16_t* wFfn1  = (bf16_t*)carve((size_t)DFFD * DD * 2);
  bf16_t* wFfn2  = (bf16_t*)carve((size_t)DD * DFFD * 2);
  bf16_t* lnx    = (bf16_t*)carve((size_t)BLD * 2);
  float*  pBuf   = (float*)carve((size_t)MROWS * 4 * DD * 4); // reused ffn1 out
  bf16_t* qS     = (bf16_t*)carve((size_t)BLD * 2);
  bf16_t* kS     = (bf16_t*)carve((size_t)BLD * 2);
  bf16_t* vT     = (bf16_t*)carve((size_t)BLD * 2);
  float*  xoS    = (float*)carve((size_t)BLD * 4);
  bf16_t* decayBF= (bf16_t*)carve((size_t)NH * LL * LL * 2);  // 32MB
  float*  retF   = (float*)carve((size_t)BLD * 4);            // GEMM f32 temp
  bf16_t* retg   = (bf16_t*)carve((size_t)BLD * 2);
  float*  x1     = (float*)carve((size_t)BLD * 4);
  bf16_t* hBF    = (bf16_t*)carve((size_t)BLD * 2);
  bf16_t* gBF    = (bf16_t*)carve((size_t)MROWS * DFFD * 2);

  const int EW_GRID = 4096;

  // 1) weights + decay -> bf16
  cvt_bf16_kernel<<<EW_GRID, 256, 0, stream>>>(proj_W, wProj, (long long)4 * DD * DD);
  cvt_bf16_kernel<<<EW_GRID, 256, 0, stream>>>(reto_W, wReto, (long long)DD * DD);
  cvt_bf16_kernel<<<EW_GRID, 256, 0, stream>>>(ffn1_W, wFfn1, (long long)DFFD * DD);
  cvt_bf16_kernel<<<EW_GRID, 256, 0, stream>>>(ffn2_W, wFfn2, (long long)DD * DFFD);
  cvt_bf16_kernel<<<EW_GRID, 256, 0, stream>>>(decay, decayBF, (long long)NH * LL * LL);

  // 2) LN1
  ln_bf16_kernel<<<MROWS, 256, 0, stream>>>(x, ln1_g, ln1_b, lnx);

  // 3) p = lnx @ projW^T + proj_b      [4096 x 4096], K=1024
  gemm_bf16_kernel<<<dim3(4 * DD / 64, MROWS / 128, 1), 256, 0, stream>>>(
      lnx, wProj, proj_b, pBuf, DD, DD, DD, 4 * DD);

  // 4) split into q/k/vT (bf16, head-major) + silu(xo) (f32)
  split_proj_kernel<<<EW_GRID, 256, 0, stream>>>(pBuf, qS, kS, vT, xoS);

  // 5) fused retention: retg = bf16( ((q@kT)*decay) @ v * silu(xo) )
  retention_kernel<<<dim3(1, LL / 128, BB * NH), 256, 0, stream>>>(
      qS, kS, vT, decayBF, xoS, retg);

  // 6) reto GEMM: retF = retg @ retoW^T + reto_b   [4096 x 1024], K=1024
  gemm_bf16_kernel<<<dim3(DD / 64, MROWS / 128, 1), 256, 0, stream>>>(
      retg, wReto, reto_b, retF, DD, DD, DD, DD);

  // 7) residual: x1 = x + retF
  add_f32_kernel<<<EW_GRID, 256, 0, stream>>>(x, retF, x1, BLD);

  // 8) LN2
  ln_bf16_kernel<<<MROWS, 256, 0, stream>>>(x1, ln2_g, ln2_b, hBF);

  // 9) ffn1: pBuf = h @ ffn1W^T + b    [4096 x 4096], K=1024
  gemm_bf16_kernel<<<dim3(DFFD / 64, MROWS / 128, 1), 256, 0, stream>>>(
      hBF, wFfn1, ffn1_b, pBuf, DD, DD, DD, DFFD);

  // 10) gelu (exact) -> bf16
  gelu_bf16_kernel<<<EW_GRID, 256, 0, stream>>>(pBuf, gBF, (long long)MROWS * DFFD);

  // 11) ffn2: retF = g @ ffn2W^T + b   [4096 x 1024], K=4096
  gemm_bf16_kernel<<<dim3(DD / 64, MROWS / 128, 1), 256, 0, stream>>>(
      gBF, wFfn2, ffn2_b, retF, DFFD, DFFD, DFFD, DD);

  // 12) out = x1 + retF
  add_f32_kernel<<<EW_GRID, 256, 0, stream>>>(x1, retF, out, BLD);
}
